// ConformerBlock_89309549953503
// MI455X (gfx1250) — compile-verified, matrix operand order checked by reading, not run on hardware
//
#include <hip/hip_runtime.h>
#include <hip/hip_bf16.h>
#include <math.h>

typedef __bf16 bf16;
typedef __bf16 v16bf __attribute__((ext_vector_type(16)));
typedef __bf16 v8bf  __attribute__((ext_vector_type(8)));
typedef float  v8f   __attribute__((ext_vector_type(8)));

constexpr int Bb  = 8;
constexpr int Ss  = 512;
constexpr int Dd  = 512;
constexpr int Hh  = 8;
constexpr int DHd = 64;
constexpr int DFF = 2048;
constexpr int Mrows = Bb * Ss;       // 4096
constexpr int NREL  = 2 * Ss - 1;    // 1023 unique relative distances

#define DEVI __device__ __forceinline__

DEVI v16bf make_frag(const bf16* p0, const bf16* p1) {
  v8bf lo = *(const v8bf*)p0;
  v8bf hi = *(const v8bf*)p1;
  v16bf f;
#pragma unroll
  for (int i = 0; i < 8; ++i) { f[i] = lo[i]; f[i + 8] = hi[i]; }
  return f;
}

DEVI v8f wmma_bf16(v16bf a, v16bf b, v8f c) {
  return __builtin_amdgcn_wmma_f32_16x16x32_bf16(false, a, false, b, (short)0, c, false, false);
}

// ---------------------------------------------------------------- utilities
__global__ void cvt_bf16_kernel(const float* __restrict__ in, bf16* __restrict__ out, int n) {
  int i = blockIdx.x * blockDim.x + threadIdx.x;
  if (i < n) out[i] = (bf16)in[i];
}

// Row LayerNorm over D=512; optional bf16 and/or f32 outputs.
__global__ __launch_bounds__(128)
void ln_kernel(const float* __restrict__ x, const float* __restrict__ g,
               const float* __restrict__ b, bf16* __restrict__ outB,
               float* __restrict__ outF) {
  const int row = blockIdx.x;
  const float* xr = x + (size_t)row * Dd;
  __shared__ float red[128];
  float s = 0.f;
  for (int i = threadIdx.x; i < Dd; i += 128) s += xr[i];
  red[threadIdx.x] = s; __syncthreads();
  for (int off = 64; off > 0; off >>= 1) {
    if (threadIdx.x < off) red[threadIdx.x] += red[threadIdx.x + off];
    __syncthreads();
  }
  float mean = red[0] / Dd;
  __syncthreads();
  float v = 0.f;
  for (int i = threadIdx.x; i < Dd; i += 128) { float d = xr[i] - mean; v += d * d; }
  red[threadIdx.x] = v; __syncthreads();
  for (int off = 64; off > 0; off >>= 1) {
    if (threadIdx.x < off) red[threadIdx.x] += red[threadIdx.x + off];
    __syncthreads();
  }
  float rstd = rsqrtf(red[0] / Dd + 1e-5f);
  for (int i = threadIdx.x; i < Dd; i += 128) {
    float t = (xr[i] - mean) * rstd * g[i] + b[i];
    if (outB) outB[(size_t)row * Dd + i] = (bf16)t;
    if (outF) outF[(size_t)row * Dd + i] = t;
  }
}

// ------------------------------------------------------ generic WMMA GEMM
// C[m,n] = epilogue( sum_k A[m,k]*W[n,k] )  (A: M x K bf16, W: N x K bf16)
// Epilogue shape is compile-time: (+bias[n]) -> SiLU? -> *scale -> (+res) -> stores
template <bool SILU, bool HASB, bool HASR, bool OF, bool OB>
__global__ __launch_bounds__(128)
void gemm_bf16_wmma(const bf16* __restrict__ A, const bf16* __restrict__ W,
                    const float* __restrict__ bias, const float* __restrict__ res,
                    float scale,
                    float* __restrict__ outF, bf16* __restrict__ outB,
                    int M, int N, int K) {
  const int lane = threadIdx.x & 31;
  const int wave = threadIdx.x >> 5;
  const int n4 = lane & 15, lh = lane >> 4;
  const int rowBase = blockIdx.y * 64 + (wave >> 1) * 32;
  const int colBase = blockIdx.x * 64 + (wave & 1) * 32;

  v8f acc[2][2] = {};
  for (int kk = 0; kk < K; kk += 32) {
    v16bf a[2], bm[2];
#pragma unroll
    for (int im = 0; im < 2; ++im) {
      int m = rowBase + im * 16 + n4; if (m >= M) m = M - 1;
      const bf16* p = A + (size_t)m * K + kk + lh * 8;
      a[im] = make_frag(p, p + 16);
      if (kk + 128 < K) __builtin_prefetch(p + 128, 0, 1);   // global_prefetch_b8, stream A
    }
#pragma unroll
    for (int jn = 0; jn < 2; ++jn) {
      int n = colBase + jn * 16 + n4; if (n >= N) n = N - 1;
      const bf16* p = W + (size_t)n * K + kk + lh * 8;
      bm[jn] = make_frag(p, p + 16);
    }
#pragma unroll
    for (int im = 0; im < 2; ++im)
#pragma unroll
      for (int jn = 0; jn < 2; ++jn)
        acc[im][jn] = wmma_bf16(a[im], bm[jn], acc[im][jn]);
  }

#pragma unroll
  for (int im = 0; im < 2; ++im)
#pragma unroll
    for (int jn = 0; jn < 2; ++jn) {
      int n = colBase + jn * 16 + n4;
      if (n >= N) continue;
      float bv = HASB ? bias[n] : 0.f;
#pragma unroll
      for (int r = 0; r < 8; ++r) {
        int m = rowBase + im * 16 + lh * 8 + r;
        if (m >= M) continue;
        float t = acc[im][jn][r] + bv;
        if (SILU) t = t / (1.f + __expf(-t));
        t *= scale;
        size_t idx = (size_t)m * N + n;
        if (HASR) t += res[idx];
        if (OF) outF[idx] = t;
        if (OB) outB[idx] = (bf16)t;
      }
    }
}

// ------------------------------------------------- small elementwise helpers
__global__ void build_quqv(const float* __restrict__ qf, const float* __restrict__ u,
                           const float* __restrict__ v, bf16* __restrict__ qu,
                           bf16* __restrict__ qv, int total) {
  int i = blockIdx.x * blockDim.x + threadIdx.x;
  if (i >= total) return;
  int d = i & (Dd - 1);
  float q = qf[i];
  qu[i] = (bf16)(q + u[d]);
  qv[i] = (bf16)(q + v[d]);
}

__global__ void transpose_v(const float* __restrict__ vf, bf16* __restrict__ vT) {
  int i = blockIdx.x * blockDim.x + threadIdx.x;   // over B*S*D
  if (i >= Mrows * Dd) return;
  int d = i & 511, s = (i >> 9) & 511, b = i >> 18;
  int h = d >> 6, dh = d & 63;
  vT[((((size_t)b * Hh + h) * DHd + dh) << 9) + s] = (bf16)vf[i];
}

// sin/cos relative-position rows for the 1023 unique distances, interleaved.
__global__ void relgen(bf16* __restrict__ Rb) {
  int i = blockIdx.x * blockDim.x + threadIdx.x;   // over 1023*256
  if (i >= NREL * (Dd / 2)) return;
  int t = i & 255, r = i >> 8;
  float rel = (float)(r - (Ss - 1));
  float div = __expf((float)(2 * t) * (-9.210340371976184f / (float)Dd));
  float ang = rel * div;
  Rb[(size_t)r * Dd + 2 * t]     = (bf16)__sinf(ang);
  Rb[(size_t)r * Dd + 2 * t + 1] = (bf16)__cosf(ang);
}

__global__ void glu_kernel(const float* __restrict__ p, float* __restrict__ c, int total) {
  int i = blockIdx.x * blockDim.x + threadIdx.x;   // over M*512
  if (i >= total) return;
  int row = i >> 9, d = i & 511;
  float a = p[(size_t)row * (2 * Dd) + d];
  float g = p[(size_t)row * (2 * Dd) + Dd + d];
  c[i] = a * (1.f / (1.f + __expf(-g)));
}

__global__ void dwconv_kernel(const float* __restrict__ c, const float* __restrict__ w,
                              const float* __restrict__ bias, const float* __restrict__ bng,
                              const float* __restrict__ bnb, bf16* __restrict__ outB) {
  int i = blockIdx.x * blockDim.x + threadIdx.x;   // over B*S*D
  if (i >= Mrows * Dd) return;
  int d = i & 511, s = (i >> 9) & 511, b = i >> 18;
  float acc = bias[d];
#pragma unroll
  for (int t = 0; t < 31; ++t) {
    int ss = s + t - 15;
    if (ss >= 0 && ss < Ss)
      acc += c[((size_t)(b * Ss + ss)) * Dd + d] * w[d * 31 + t];
  }
  float bn = acc * rsqrtf(1.f + 1e-5f) * bng[d] + bnb[d];
  float y = bn / (1.f + __expf(-bn));
  outB[i] = (bf16)y;
}

// ---------------------------------------------------- flash rel-pos attention
// Transposed score layout: keys = WMMA M, queries = WMMA N.
// Per 16-query tile: stream keys in 32-chunks; content via k-rows x qu,
// position via rp band-rows x qv (+ per-wave LDS diagonal gather),
// online softmax (one shfl_xor(16)), probs feed P.V WMMA with zero shuffles.
__global__ __launch_bounds__(128)
void flash_relattn(const bf16* __restrict__ qu, const bf16* __restrict__ qv,
                   const bf16* __restrict__ kb, const bf16* __restrict__ vT,
                   const bf16* __restrict__ rp, bf16* __restrict__ outB) {
  __shared__ float lds[4][32 * 16];
  const int wave = threadIdx.x >> 5;
  const int lane = threadIdx.x & 31;
  const int n = lane & 15, lh = lane >> 4;
  const int bh = blockIdx.y;
  const int b = bh >> 3, h = bh & 7;
  const int i0 = (blockIdx.x * 4 + wave) * 16;
  float* P = lds[wave];

  // B-fragments (queries as columns) for q+u and q+v, K = DH in 2 steps of 32
  v16bf bqu[2], bqv[2];
  const size_t qbase = ((size_t)(b * Ss + i0 + n)) * Dd + h * DHd + lh * 8;
#pragma unroll
  for (int kk = 0; kk < 2; ++kk) {
    const bf16* p0 = qu + qbase + kk * 32;
    bqu[kk] = make_frag(p0, p0 + 16);
    const bf16* p1 = qv + qbase + kk * 32;
    bqv[kk] = make_frag(p1, p1 + 16);
  }

  float mrow = -1e30f, lrow = 0.f;
  v8f o[4] = {};

  for (int j0 = 0; j0 < Ss; j0 += 32) {
    // prefetch next key/value chunk while this one computes
    if (j0 + 32 < Ss) {
      const bf16* kpre = kb + ((size_t)(b * Ss + j0 + 32 + n)) * Dd + h * DHd + lh * 8;
      __builtin_prefetch(kpre, 0, 1);
      const bf16* vpre = vT + (((size_t)(b * Hh + h) * DHd + n)) * Ss + j0 + 32 + lh * 8;
      __builtin_prefetch(vpre, 0, 1);
    }
    float pr[2][8];
#pragma unroll
    for (int jt = 0; jt < 2; ++jt) {
      const int j0s = j0 + jt * 16;
      // content scores^T: A = key rows (16x64), B = qu (64x16)
      v8f sc = {};
#pragma unroll
      for (int kk = 0; kk < 2; ++kk) {
        const bf16* kp = kb + ((size_t)(b * Ss + j0s + n)) * Dd + h * DHd + kk * 32 + lh * 8;
        v16bf ak = make_frag(kp, kp + 16);
        sc = wmma_bf16(ak, bqu[kk], sc);
      }
      // position band: rows r = (j - i) + 511; for this tile r = rb0 + (mj - n + 15)
      const int rb0 = j0s - i0 + 496;
      v8f pa[2] = {};
#pragma unroll
      for (int t = 0; t < 2; ++t) {
#pragma unroll
        for (int kk = 0; kk < 2; ++kk) {
          int rr = rb0 + t * 16 + n;
          rr = rr < 0 ? 0 : (rr > NREL - 1 ? NREL - 1 : rr);
          const bf16* rpp = rp + (size_t)rr * Dd + h * DHd + kk * 32 + lh * 8;
          v16bf ar = make_frag(rpp, rpp + 16);
          pa[t] = wmma_bf16(ar, bqv[kk], pa[t]);
        }
      }
      // dump band to per-wave LDS (rows = band index, cols = query)
#pragma unroll
      for (int t = 0; t < 2; ++t)
#pragma unroll
        for (int r = 0; r < 8; ++r)
          P[(t * 16 + lh * 8 + r) * 16 + n] = pa[t][r];
      // diagonal gather + scale by DH^-0.5 (same-wave LDS ops are in-order)
#pragma unroll
      for (int r = 0; r < 8; ++r) {
        int mj = lh * 8 + r;
        float sv = sc[r] + P[(mj - n + 15) * 16 + n];
        pr[jt][r] = sv * 0.125f;
      }
    }
    // ---- online softmax over this 32-key chunk
    float cmax = -1e30f;
#pragma unroll
    for (int jt = 0; jt < 2; ++jt)
#pragma unroll
      for (int r = 0; r < 8; ++r) cmax = fmaxf(cmax, pr[jt][r]);
    cmax = fmaxf(cmax, __shfl_xor(cmax, 16, 32));
    float mnew = fmaxf(mrow, cmax);
    float alpha = __expf(mrow - mnew);
    float psum = 0.f;
#pragma unroll
    for (int jt = 0; jt < 2; ++jt)
#pragma unroll
      for (int r = 0; r < 8; ++r) { float e = __expf(pr[jt][r] - mnew); pr[jt][r] = e; psum += e; }
    psum += __shfl_xor(psum, 16, 32);
    lrow = lrow * alpha + psum;
    mrow = mnew;
#pragma unroll
    for (int td = 0; td < 4; ++td)
#pragma unroll
      for (int r = 0; r < 8; ++r) o[td][r] *= alpha;
    // probs already sit in B-fragment layout: e<8 -> keys lh*8+e (subtile 0),
    // e>=8 -> keys 16+lh*8+(e-8) (subtile 1)
    v16bf pb;
#pragma unroll
    for (int e = 0; e < 8; ++e) { pb[e] = (bf16)pr[0][e]; pb[e + 8] = (bf16)pr[1][e]; }
    // O^T += vT_chunk (16dh x 32keys) x P (32keys x 16q)
#pragma unroll
    for (int td = 0; td < 4; ++td) {
      const bf16* vp = vT + (((size_t)(b * Hh + h) * DHd + td * 16 + n)) * Ss + j0 + lh * 8;
      v16bf av = make_frag(vp, vp + 16);
      o[td] = wmma_bf16(av, pb, o[td]);
    }
  }
  const float inv = 1.f / lrow;
#pragma unroll
  for (int td = 0; td < 4; ++td)
#pragma unroll
    for (int r = 0; r < 8; ++r) {
      int dh = td * 16 + lh * 8 + r;
      int i = i0 + n;
      outB[((size_t)(b * Ss + i)) * Dd + h * DHd + dh] = (bf16)(o[td][r] * inv);
    }
}

// ---------------------------------------------------------------- launcher
extern "C" void kernel_launch(void* const* d_in, const int* in_sizes, int n_in,
                              void* d_out, int out_size, void* d_ws, size_t ws_size,
                              hipStream_t stream) {
  (void)in_sizes; (void)n_in; (void)out_size; (void)ws_size;
  const float* x        = (const float*)d_in[0];
  const float* ff1_ln_g = (const float*)d_in[1];
  const float* ff1_ln_b = (const float*)d_in[2];
  const float* ff1_w1   = (const float*)d_in[3];
  const float* ff1_b1   = (const float*)d_in[4];
  const float* ff1_w2   = (const float*)d_in[5];
  const float* ff1_b2   = (const float*)d_in[6];
  const float* ff2_ln_g = (const float*)d_in[7];
  const float* ff2_ln_b = (const float*)d_in[8];
  const float* ff2_w1   = (const float*)d_in[9];
  const float* ff2_b1   = (const float*)d_in[10];
  const float* ff2_w2   = (const float*)d_in[11];
  const float* ff2_b2   = (const float*)d_in[12];
  const float* attn_ln_g = (const float*)d_in[13];
  const float* attn_ln_b = (const float*)d_in[14];
  const float* wq = (const float*)d_in[15];
  const float* wk = (const float*)d_in[16];
  const float* wv = (const float*)d_in[17];
  const float* wo = (const float*)d_in[18];
  const float* wr = (const float*)d_in[19];
  const float* bq = (const float*)d_in[20];
  const float* bk = (const float*)d_in[21];
  const float* bv = (const float*)d_in[22];
  const float* bo = (const float*)d_in[23];
  const float* uu = (const float*)d_in[24];
  const float* vv = (const float*)d_in[25];
  const float* conv_ln_g = (const float*)d_in[26];
  const float* conv_ln_b = (const float*)d_in[27];
  const float* pw1_w = (const float*)d_in[28];
  const float* pw1_b = (const float*)d_in[29];
  const float* dw_w  = (const float*)d_in[30];
  const float* dw_b  = (const float*)d_in[31];
  const float* bn_g  = (const float*)d_in[32];
  const float* bn_b  = (const float*)d_in[33];
  const float* pw2_w = (const float*)d_in[34];
  const float* pw2_b = (const float*)d_in[35];
  const float* fin_ln_g = (const float*)d_in[36];
  const float* fin_ln_b = (const float*)d_in[37];

  size_t off = 0;
  auto alloc = [&](size_t bytes) -> void* {
    void* p = (char*)d_ws + off;
    off += (bytes + 255) & ~(size_t)255;
    return p;
  };
  // bf16 weights
  bf16* wb_ff1w1 = (bf16*)alloc((size_t)DFF * Dd * 2);
  bf16* wb_ff1w2 = (bf16*)alloc((size_t)Dd * DFF * 2);
  bf16* wb_ff2w1 = (bf16*)alloc((size_t)DFF * Dd * 2);
  bf16* wb_ff2w2 = (bf16*)alloc((size_t)Dd * DFF * 2);
  bf16* wb_wq = (bf16*)alloc((size_t)Dd * Dd * 2);
  bf16* wb_wk = (bf16*)alloc((size_t)Dd * Dd * 2);
  bf16* wb_wv = (bf16*)alloc((size_t)Dd * Dd * 2);
  bf16* wb_wo = (bf16*)alloc((size_t)Dd * Dd * 2);
  bf16* wb_wr = (bf16*)alloc((size_t)Dd * Dd * 2);
  bf16* wb_pw1 = (bf16*)alloc((size_t)2 * Dd * Dd * 2);
  bf16* wb_pw2 = (bf16*)alloc((size_t)Dd * Dd * 2);
  // activations
  float* xb   = (float*)alloc((size_t)Mrows * Dd * 4);       // running residual stream
  bf16* y_bf  = (bf16*)alloc((size_t)Mrows * Dd * 2);        // post-LN bf16
  void* big16 = alloc((size_t)Mrows * DFF * 2);              // h_bf (bf16) / conv p (f32)
  float* qf   = (float*)alloc((size_t)Mrows * Dd * 4);       // q f32 -> v f32 -> conv glu
  bf16* kb    = (bf16*)alloc((size_t)Mrows * Dd * 2);
  bf16* qub   = (bf16*)alloc((size_t)Mrows * Dd * 2);
  bf16* qvb   = (bf16*)alloc((size_t)Mrows * Dd * 2);
  bf16* vTb   = (bf16*)alloc((size_t)Mrows * Dd * 2);
  bf16* Rb    = (bf16*)alloc((size_t)NREL * Dd * 2);
  bf16* rpb   = (bf16*)alloc((size_t)NREL * Dd * 2);
  bf16* attn_bf = (bf16*)alloc((size_t)Mrows * Dd * 2);

  auto CVT = [&](const float* src, bf16* dst, int n) {
    cvt_bf16_kernel<<<(n + 255) / 256, 256, 0, stream>>>(src, dst, n);
  };
  dim3 blk(128);
  // Epilogue variants (compile-time): plain->bf16, bias+silu->bf16, bias->f32,
  // bias->bf16, bias+res->f32
  auto GEMM_B = [&](const bf16* A, const bf16* W, bf16* oB, int M, int N, int K) {
    dim3 g((N + 63) / 64, (M + 63) / 64);
    gemm_bf16_wmma<false, false, false, false, true><<<g, blk, 0, stream>>>(
        A, W, nullptr, nullptr, 1.f, nullptr, oB, M, N, K);
  };
  auto GEMM_BiasSiluB = [&](const bf16* A, const bf16* W, const float* bias, bf16* oB,
                            int M, int N, int K) {
    dim3 g((N + 63) / 64, (M + 63) / 64);
    gemm_bf16_wmma<true, true, false, false, true><<<g, blk, 0, stream>>>(
        A, W, bias, nullptr, 1.f, nullptr, oB, M, N, K);
  };
  auto GEMM_BiasF = [&](const bf16* A, const bf16* W, const float* bias, float* oF,
                        int M, int N, int K) {
    dim3 g((N + 63) / 64, (M + 63) / 64);
    gemm_bf16_wmma<false, true, false, true, false><<<g, blk, 0, stream>>>(
        A, W, bias, nullptr, 1.f, oF, nullptr, M, N, K);
  };
  auto GEMM_BiasB = [&](const bf16* A, const bf16* W, const float* bias, bf16* oB,
                        int M, int N, int K) {
    dim3 g((N + 63) / 64, (M + 63) / 64);
    gemm_bf16_wmma<false, true, false, false, true><<<g, blk, 0, stream>>>(
        A, W, bias, nullptr, 1.f, nullptr, oB, M, N, K);
  };
  auto GEMM_BiasResF = [&](const bf16* A, const bf16* W, const float* bias,
                           const float* res, float scale, float* oF, int M, int N, int K) {
    dim3 g((N + 63) / 64, (M + 63) / 64);
    gemm_bf16_wmma<false, true, true, true, false><<<g, blk, 0, stream>>>(
        A, W, bias, res, scale, oF, nullptr, M, N, K);
  };

  // weight conversions (deterministic per call)
  CVT(ff1_w1, wb_ff1w1, DFF * Dd);
  CVT(ff1_w2, wb_ff1w2, Dd * DFF);
  CVT(ff2_w1, wb_ff2w1, DFF * Dd);
  CVT(ff2_w2, wb_ff2w2, Dd * DFF);
  CVT(wq, wb_wq, Dd * Dd);
  CVT(wk, wb_wk, Dd * Dd);
  CVT(wv, wb_wv, Dd * Dd);
  CVT(wo, wb_wo, Dd * Dd);
  CVT(wr, wb_wr, Dd * Dd);
  CVT(pw1_w, wb_pw1, 2 * Dd * Dd);
  CVT(pw2_w, wb_pw2, Dd * Dd);

  // relative-position projection: 1023 unique distances only
  relgen<<<(NREL * (Dd / 2) + 255) / 256, 256, 0, stream>>>(Rb);
  GEMM_B(Rb, wb_wr, rpb, NREL, Dd, Dd);

  // ---- FFN1 (macaron, 0.5x)
  ln_kernel<<<Mrows, 128, 0, stream>>>(x, ff1_ln_g, ff1_ln_b, y_bf, nullptr);
  GEMM_BiasSiluB(y_bf, wb_ff1w1, ff1_b1, (bf16*)big16, Mrows, DFF, Dd);
  GEMM_BiasResF((bf16*)big16, wb_ff1w2, ff1_b2, x, 0.5f, xb, Mrows, Dd, DFF);

  // ---- Attention
  ln_kernel<<<Mrows, 128, 0, stream>>>(xb, attn_ln_g, attn_ln_b, y_bf, nullptr);
  GEMM_BiasF(y_bf, wb_wq, bq, qf, Mrows, Dd, Dd);
  GEMM_BiasB(y_bf, wb_wk, bk, kb, Mrows, Dd, Dd);
  build_quqv<<<(Mrows * Dd + 255) / 256, 256, 0, stream>>>(qf, uu, vv, qub, qvb, Mrows * Dd);
  GEMM_BiasF(y_bf, wb_wv, bv, qf, Mrows, Dd, Dd);   // reuse qf as v
  transpose_v<<<(Mrows * Dd + 255) / 256, 256, 0, stream>>>(qf, vTb);
  flash_relattn<<<dim3(Ss / 64, Bb * Hh), 128, 0, stream>>>(qub, qvb, kb, vTb, rpb, attn_bf);
  GEMM_BiasResF(attn_bf, wb_wo, bo, xb, 1.f, xb, Mrows, Dd, Dd);

  // ---- Conv module
  ln_kernel<<<Mrows, 128, 0, stream>>>(xb, conv_ln_g, conv_ln_b, y_bf, nullptr);
  GEMM_BiasF(y_bf, wb_pw1, pw1_b, (float*)big16, Mrows, 2 * Dd, Dd);
  glu_kernel<<<(Mrows * Dd + 255) / 256, 256, 0, stream>>>((const float*)big16, qf, Mrows * Dd);
  dwconv_kernel<<<(Mrows * Dd + 255) / 256, 256, 0, stream>>>(qf, dw_w, dw_b, bn_g, bn_b, y_bf);
  GEMM_BiasResF(y_bf, wb_pw2, pw2_b, xb, 1.f, xb, Mrows, Dd, Dd);

  // ---- FFN2
  ln_kernel<<<Mrows, 128, 0, stream>>>(xb, ff2_ln_g, ff2_ln_b, y_bf, nullptr);
  GEMM_BiasSiluB(y_bf, wb_ff2w1, ff2_b1, (bf16*)big16, Mrows, DFF, Dd);
  GEMM_BiasResF((bf16*)big16, wb_ff2w2, ff2_b2, xb, 0.5f, xb, Mrows, Dd, DFF);

  // ---- final LayerNorm -> f32 output
  ln_kernel<<<Mrows, 128, 0, stream>>>(xb, fin_ln_g, fin_ln_b, nullptr, (float*)d_out);
}
